// DotProductAttention_35725537968478
// MI455X (gfx1250) — matrix operand order repaired and checked by reading, not compiled, not run on hardware
//
#include <hip/hip_runtime.h>
#include <hip/hip_bf16.h>

// ---- CDNA5 WMMA types ----
typedef __attribute__((ext_vector_type(16))) __bf16 v16bf;
typedef __attribute__((ext_vector_type(8)))  __bf16 v8bf;
typedef __attribute__((ext_vector_type(8)))  float  v8f;

// Problem constants (match reference setup_inputs)
#define QLEN   256
#define KVLEN  256
#define DHEAD  64
#define NWIN   64
#define WAVES  4          // 128 threads / block, wave32
#define ROWS_PER_WAVE 16  // one WMMA M-tile per wave

// ---------------------------------------------------------------------------
// Fragment builders (CDNA5 ISA §7.12.2, 16-bit layouts)
// A (16x32, MxK): lane l: M=l&15, elems 0..7 = K[(l>>4)*8 .. +7], 8..15 = +16
// B (32x16, KxN): lane l: N=l&15, elems 0..7 = K[(l>>4)*8 .. +7], 8..15 = +16
// ---------------------------------------------------------------------------

// A fragment from row-major f32 global (K offset folded into src)
__device__ __forceinline__ v16bf frag_A_f32(const float* __restrict__ src, int ld, int lane) {
    const int row = lane & 15, hf = lane >> 4;
    const float* p = src + row * ld + hf * 8;
    v16bf a;
#pragma unroll
    for (int j = 0; j < 8; ++j) { a[j] = (__bf16)p[j]; a[8 + j] = (__bf16)p[16 + j]; }
    return a;
}

// A fragment from row-major bf16 in LDS
__device__ __forceinline__ v16bf frag_A_lds(const __bf16* src, int ld, int lane) {
    const int row = lane & 15, hf = lane >> 4;
    const __bf16* p = src + row * ld + hf * 8;
    v16bf a;
#pragma unroll
    for (int j = 0; j < 8; ++j) { a[j] = p[j]; a[8 + j] = p[16 + j]; }
    return a;
}

// B fragment where B[k][n] = src[n*ld + k]  (elems 0..7 = K[hf*8..+7], 8..15 = +16)
// Used for K^T (K row-major) and for V via its transposed LDS copy.
__device__ __forceinline__ v16bf frag_B_rows_lds(const __bf16* src, int ld, int lane) {
    const int n = lane & 15, kb = (lane >> 4) * 8;
    const __bf16* p = src + n * ld + kb;
    v16bf b;
#pragma unroll
    for (int e = 0; e < 8; ++e) { b[e] = p[e]; b[8 + e] = p[16 + e]; }
    return b;
}

// ---------------------------------------------------------------------------

__global__ void __launch_bounds__(WAVES * 32)
swin_attention_kernel(const float* __restrict__ Qg,
                      const float* __restrict__ Kg,
                      const float* __restrict__ Vg,
                      const int*   __restrict__ valid_lens,
                      const float* __restrict__ Wmask,
                      float* __restrict__ Out) {
    // ldsA: first holds K as bf16 row-major [kv][d]; after the score GEMM it is
    // reused as the P (probabilities) scratch [wave][16][kv]. Both are 16384 bf16.
    __shared__ __bf16 ldsA[KVLEN * DHEAD];        // 32 KB
    __shared__ __bf16 ldsVt[DHEAD * KVLEN];       // 32 KB, V transposed [d][kv]

    const int tid  = threadIdx.x;
    const int wave = tid >> 5;
    const int lane = tid & 31;
    const int hf   = lane >> 4;
    const int nl   = lane & 15;

    const int tiles_per_b = QLEN / (WAVES * ROWS_PER_WAVE);   // 4
    const int bb    = blockIdx.x / tiles_per_b;
    const int qtile = blockIdx.x % tiles_per_b;
    const int qbase = qtile * (WAVES * ROWS_PER_WAVE) + wave * ROWS_PER_WAVE;

    const int win  = (bb >> 3) & (NWIN - 1);                  // (bb / NUM_HEADS) % W
    const int vlen = valid_lens[bb];

    const float* Qp = Qg + ((size_t)bb * QLEN + qbase) * DHEAD;
    const float* Kp = Kg + (size_t)bb * KVLEN * DHEAD;
    const float* Vp = Vg + (size_t)bb * KVLEN * DHEAD;
    const float* Mp = Wmask + (size_t)win * QLEN * KVLEN;

    // ---- cooperative stage: K -> ldsA (row-major bf16), V -> ldsVt (transposed bf16)
    // 16384 elements each; 128 threads x 16 iters x 8 elems, fully coalesced b128 loads.
#pragma unroll
    for (int j = 0; j < (KVLEN * DHEAD) / (WAVES * 32 * 8); ++j) {   // 16 iters
        const int base = (j * (WAVES * 32) + tid) * 8;
        // K: straight copy with convert
        float4 k0 = *(const float4*)(Kp + base);
        float4 k1 = *(const float4*)(Kp + base + 4);
        v8bf kb;
        kb[0] = (__bf16)k0.x; kb[1] = (__bf16)k0.y; kb[2] = (__bf16)k0.z; kb[3] = (__bf16)k0.w;
        kb[4] = (__bf16)k1.x; kb[5] = (__bf16)k1.y; kb[6] = (__bf16)k1.z; kb[7] = (__bf16)k1.w;
        *(v8bf*)(ldsA + base) = kb;
        // V: transpose while staging: (kv,d) -> ldsVt[d*KVLEN + kv]
        const int kvr = base >> 6;        // DHEAD = 64
        const int dc  = base & (DHEAD - 1);
        float4 v0 = *(const float4*)(Vp + base);
        float4 v1 = *(const float4*)(Vp + base + 4);
        ldsVt[(dc + 0) * KVLEN + kvr] = (__bf16)v0.x;
        ldsVt[(dc + 1) * KVLEN + kvr] = (__bf16)v0.y;
        ldsVt[(dc + 2) * KVLEN + kvr] = (__bf16)v0.z;
        ldsVt[(dc + 3) * KVLEN + kvr] = (__bf16)v0.w;
        ldsVt[(dc + 4) * KVLEN + kvr] = (__bf16)v1.x;
        ldsVt[(dc + 5) * KVLEN + kvr] = (__bf16)v1.y;
        ldsVt[(dc + 6) * KVLEN + kvr] = (__bf16)v1.z;
        ldsVt[(dc + 7) * KVLEN + kvr] = (__bf16)v1.w;
    }
    __syncthreads();

    // ---- Q fragments (16 x 64 -> two K=32 chunks), f32 -> bf16 once per wave ----
    const v16bf aq0 = frag_A_f32(Qp, DHEAD, lane);
    const v16bf aq1 = frag_A_f32(Qp + 32, DHEAD, lane);

    // ---- scores: S = Q K^T, 16 x 256 strip in registers (K from LDS) ----
    v8f acc[16];
#pragma unroll
    for (int t = 0; t < 16; ++t) {
        const __bf16* kt = ldsA + (t * 16) * DHEAD;
        v16bf b0 = frag_B_rows_lds(kt, DHEAD, lane);
        v16bf b1 = frag_B_rows_lds(kt + 32, DHEAD, lane);
        v8f c = {};
        c = __builtin_amdgcn_wmma_f32_16x16x32_bf16(false, aq0, false, b0, (short)0, c, false, false);
        c = __builtin_amdgcn_wmma_f32_16x16x32_bf16(false, aq1, false, b1, (short)0, c, false, false);
        acc[t] = c;
    }

    // ---- scale + window mask + valid-length mask + row softmax (in registers) ----
    const float scale = 0.125f;  // 1/sqrt(64)
#pragma unroll
    for (int r = 0; r < 8; ++r) {
        const int rowl = r + 8 * hf;
        const float* mrow = Mp + (size_t)(qbase + rowl) * KVLEN;

        float mx = -3.0e38f;
#pragma unroll
        for (int t = 0; t < 16; ++t) {
            const int n = t * 16 + nl;
            float x = acc[t][r] * scale + mrow[n];
            x = (n < vlen) ? x : -1.0e6f;                  // matches reference masking
            acc[t][r] = x;
            mx = fmaxf(mx, x);
        }
        mx = fmaxf(mx, __shfl_xor(mx, 1, 32));
        mx = fmaxf(mx, __shfl_xor(mx, 2, 32));
        mx = fmaxf(mx, __shfl_xor(mx, 4, 32));
        mx = fmaxf(mx, __shfl_xor(mx, 8, 32));

        float sum = 0.0f;
#pragma unroll
        for (int t = 0; t < 16; ++t) {
            float e = __expf(acc[t][r] - mx);
            acc[t][r] = e;
            sum += e;
        }
        sum += __shfl_xor(sum, 1, 32);
        sum += __shfl_xor(sum, 2, 32);
        sum += __shfl_xor(sum, 4, 32);
        sum += __shfl_xor(sum, 8, 32);
        const float inv = 1.0f / sum;
#pragma unroll
        for (int t = 0; t < 16; ++t) acc[t][r] *= inv;     // final probabilities
    }

    // All waves are done reading K from ldsA -> safe to reuse it for P.
    __syncthreads();

    __bf16* Prow = ldsA + wave * ROWS_PER_WAVE * KVLEN;
#pragma unroll
    for (int r = 0; r < 8; ++r) {
        const int rowl = r + 8 * hf;
#pragma unroll
        for (int t = 0; t < 16; ++t)
            Prow[rowl * KVLEN + t * 16 + nl] = (__bf16)acc[t][r];
    }
    __syncthreads();   // (cheap; guarantees P visible before re-layout reads)

    // ---- O = P V : 8 K-chunks of 32 over kv, 4 N-tiles of 16 over d ----
    v8f oacc[4];
#pragma unroll
    for (int nt = 0; nt < 4; ++nt) oacc[nt] = (v8f){};

#pragma unroll
    for (int kc = 0; kc < 8; ++kc) {
        v16bf ap = frag_A_lds(Prow + kc * 32, KVLEN, lane);
#pragma unroll
        for (int nt = 0; nt < 4; ++nt) {
            // B[k][n] = V[kc*32+k][nt*16+n] = ldsVt[(nt*16+n)*KVLEN + kc*32 + k]
            v16bf bv = frag_B_rows_lds(ldsVt + (nt * 16) * KVLEN + kc * 32, KVLEN, lane);
            oacc[nt] = __builtin_amdgcn_wmma_f32_16x16x32_bf16(false, ap, false, bv,
                                                               (short)0, oacc[nt], false, false);
        }
    }

    // ---- store output (f32) ----
#pragma unroll
    for (int r = 0; r < 8; ++r) {
        const int rowl = r + 8 * hf;
        float* orow = Out + ((size_t)bb * QLEN + qbase + rowl) * DHEAD;
#pragma unroll
        for (int nt = 0; nt < 4; ++nt)
            orow[nt * 16 + nl] = oacc[nt][r];
    }
}

extern "C" void kernel_launch(void* const* d_in, const int* in_sizes, int n_in,
                              void* d_out, int out_size, void* d_ws, size_t ws_size,
                              hipStream_t stream) {
    (void)n_in; (void)out_size; (void)d_ws; (void)ws_size;
    const float* q  = (const float*)d_in[0];
    const float* k  = (const float*)d_in[1];
    const float* v  = (const float*)d_in[2];
    const int*   vl = (const int*)d_in[3];
    const float* wm = (const float*)d_in[4];
    float* out = (float*)d_out;

    const int n = in_sizes[3];                               // 2048 batch-heads
    const int tiles_per_b = QLEN / (WAVES * ROWS_PER_WAVE);  // 4
    dim3 grid(n * tiles_per_b);
    dim3 block(WAVES * 32);
    swin_attention_kernel<<<grid, block, 0, stream>>>(q, k, v, vl, wm, out);
}
